// PyGEGNNDecoder_6236292514324
// MI455X (gfx1250) — matrix operand order repaired and checked by reading, not compiled, not run on hardware
//
#include <hip/hip_runtime.h>
#include <hip/hip_bf16.h>

typedef __attribute__((ext_vector_type(16))) _Float16 v16h;
typedef __attribute__((ext_vector_type(8)))  float    v8f;

#define N_NODES 16384
#define N_EDGES 262144
#define HDIM 128
#define NWAVES 4

// SiLU with fast hardware reciprocal: v_exp + v_add + v_rcp + v_mul
// (avoids the IEEE v_div_scale/newton sequence the plain division emits).
__device__ __forceinline__ float silu_f(float x) {
    return x * __builtin_amdgcn_rcpf(1.0f + __expf(-x));
}

// A fragment (16x32 f16, M rows in LDS row-major, stride in halves).
// Lane L<16 holds row L, K = {k0..k0+7, k0+16..k0+23}; lane L>=16 holds row
// L-16, K = {k0+8..k0+15, k0+24..k0+31}  (ISA 7.12.2 16-bit A layout).
__device__ __forceinline__ v16h load_A_lds(const _Float16* buf, int lane, int k0, int stride) {
    int r  = lane & 15;
    int h8 = (lane >> 4) << 3;
    const _Float16* p = buf + r * stride + k0 + h8;
    v16h a;
    ((uint4*)&a)[0] = *(const uint4*)p;
    ((uint4*)&a)[1] = *(const uint4*)(p + 16);
    return a;
}

// B fragment (32x16 f16) from transposed weights Wt[outN][inK] in global (L2).
// Lane L holds column n0+(L&15); K: lanes<16 -> {k0..k0+7, k0+16..k0+23},
// lanes>=16 -> {k0+8..k0+15, k0+24..k0+31}.
__device__ __forceinline__ v16h load_B_glb(const _Float16* wt, int lane, int n0, int k0, int K) {
    int c  = lane & 15;
    int kh = (lane >> 4) << 3;
    const _Float16* p = wt + (size_t)(n0 + c) * K + k0 + kh;
    v16h b;
    ((uint4*)&b)[0] = *(const uint4*)p;
    ((uint4*)&b)[1] = *(const uint4*)(p + 16);
    return b;
}

// ---------------- weight prep: f32 [K][O] -> f16 transposed [O][K] -----------
__global__ void transpose_f16_kernel(_Float16* __restrict__ dst, const float* __restrict__ src,
                                     int O, int K, int ld) {
    int i = blockIdx.x * blockDim.x + threadIdx.x;
    if (i >= O * K) return;
    int o = i / K, k = i - o * K;
    dst[i] = (_Float16)src[(size_t)k * ld + o];
}

__global__ void transpose_pad_f16_kernel(_Float16* __restrict__ dst, const float* __restrict__ src,
                                         int O, int K, int Ksrc, int ld) {
    int i = blockIdx.x * blockDim.x + threadIdx.x;
    if (i >= O * K) return;
    int o = i / K, k = i - o * K;
    dst[i] = (k < Ksrc) ? (_Float16)src[(size_t)k * ld + o] : (_Float16)0.f;
}

// ---------------- latent injection: h = MLP([atom_onehot | z_rep]) -----------
__global__ __launch_bounds__(NWAVES * 32) void egnn_inject_kernel(
    const float* __restrict__ z, const float* __restrict__ atom,
    const _Float16* __restrict__ WtI1, const float* __restrict__ b1,
    const _Float16* __restrict__ WtI2, const float* __restrict__ b2,
    const _Float16* __restrict__ WtI3, const float* __restrict__ b3,
    float* __restrict__ h32, _Float16* __restrict__ h16) {
    __shared__ __align__(16) _Float16 sX[NWAVES][16][96];
    __shared__ __align__(16) _Float16 sY[NWAVES][16][256];
    __shared__ __align__(16) _Float16 sZ[NWAVES][16][HDIM];

    const int wave = threadIdx.x >> 5;
    const int lane = threadIdx.x & 31;
    const int n0   = (blockIdx.x * NWAVES + wave) << 4;
    const int c16  = lane & 15;
    const int rb   = (lane >> 4) << 3;

    if (lane < 16) {
        int n = n0 + lane;
        int mol = n >> 6;                      // 64 atoms / molecule
        for (int j = 0; j < 16; ++j) sX[wave][lane][j]      = (_Float16)atom[(size_t)n * 16 + j];
        for (int j = 0; j < 64; ++j) sX[wave][lane][16 + j] = (_Float16)z[(size_t)mol * 64 + j];
        for (int j = 0; j < 16; ++j) sX[wave][lane][80 + j] = (_Float16)0.f;
    }
    __builtin_amdgcn_wave_barrier();

    // GEMM1: [16,96] x [96,256] -> SiLU -> sY  (two nt-tiles in flight)
    for (int nt = 0; nt < 16; nt += 2) {
        v8f acc0, acc1;
        float bv0 = b1[nt * 16 + c16];
        float bv1 = b1[nt * 16 + 16 + c16];
        #pragma unroll
        for (int i = 0; i < 8; ++i) { acc0[i] = bv0; acc1[i] = bv1; }
        #pragma unroll
        for (int kt = 0; kt < 3; ++kt) {
            v16h a  = load_A_lds(&sX[wave][0][0], lane, kt * 32, 96);
            v16h b0 = load_B_glb(WtI1, lane, nt * 16,      kt * 32, 96);
            v16h b1v = load_B_glb(WtI1, lane, nt * 16 + 16, kt * 32, 96);
            acc0 = __builtin_amdgcn_wmma_f32_16x16x32_f16(false, a, false, b0,  (short)0, acc0, false, false);
            acc1 = __builtin_amdgcn_wmma_f32_16x16x32_f16(false, a, false, b1v, (short)0, acc1, false, false);
        }
        #pragma unroll
        for (int i = 0; i < 8; ++i) {
            sY[wave][rb + i][nt * 16 + c16]      = (_Float16)silu_f(acc0[i]);
            sY[wave][rb + i][nt * 16 + 16 + c16] = (_Float16)silu_f(acc1[i]);
        }
    }
    __builtin_amdgcn_wave_barrier();

    // GEMM2: [16,256] x [256,128] -> SiLU -> sZ
    for (int nt = 0; nt < 8; nt += 2) {
        v8f acc0, acc1;
        float bv0 = b2[nt * 16 + c16];
        float bv1 = b2[nt * 16 + 16 + c16];
        #pragma unroll
        for (int i = 0; i < 8; ++i) { acc0[i] = bv0; acc1[i] = bv1; }
        #pragma unroll
        for (int kt = 0; kt < 8; ++kt) {
            v16h a  = load_A_lds(&sY[wave][0][0], lane, kt * 32, 256);
            v16h b0 = load_B_glb(WtI2, lane, nt * 16,      kt * 32, 256);
            v16h b1v = load_B_glb(WtI2, lane, nt * 16 + 16, kt * 32, 256);
            acc0 = __builtin_amdgcn_wmma_f32_16x16x32_f16(false, a, false, b0,  (short)0, acc0, false, false);
            acc1 = __builtin_amdgcn_wmma_f32_16x16x32_f16(false, a, false, b1v, (short)0, acc1, false, false);
        }
        #pragma unroll
        for (int i = 0; i < 8; ++i) {
            sZ[wave][rb + i][nt * 16 + c16]      = (_Float16)silu_f(acc0[i]);
            sZ[wave][rb + i][nt * 16 + 16 + c16] = (_Float16)silu_f(acc1[i]);
        }
    }
    __builtin_amdgcn_wave_barrier();

    // GEMM3: [16,128] x [128,128] -> h (f32 + f16 copies)
    for (int nt = 0; nt < 8; nt += 2) {
        v8f acc0, acc1;
        float bv0 = b3[nt * 16 + c16];
        float bv1 = b3[nt * 16 + 16 + c16];
        #pragma unroll
        for (int i = 0; i < 8; ++i) { acc0[i] = bv0; acc1[i] = bv1; }
        #pragma unroll
        for (int kt = 0; kt < 4; ++kt) {
            v16h a  = load_A_lds(&sZ[wave][0][0], lane, kt * 32, HDIM);
            v16h b0 = load_B_glb(WtI3, lane, nt * 16,      kt * 32, HDIM);
            v16h b1v = load_B_glb(WtI3, lane, nt * 16 + 16, kt * 32, HDIM);
            acc0 = __builtin_amdgcn_wmma_f32_16x16x32_f16(false, a, false, b0,  (short)0, acc0, false, false);
            acc1 = __builtin_amdgcn_wmma_f32_16x16x32_f16(false, a, false, b1v, (short)0, acc1, false, false);
        }
        #pragma unroll
        for (int i = 0; i < 8; ++i) {
            int n = n0 + rb + i;
            int col0 = nt * 16 + c16, col1 = col0 + 16;
            h32[(size_t)n * HDIM + col0] = acc0[i];
            h16[(size_t)n * HDIM + col0] = (_Float16)acc0[i];
            h32[(size_t)n * HDIM + col1] = acc1[i];
            h16[(size_t)n * HDIM + col1] = (_Float16)acc1[i];
        }
    }
}

// ---------------- edge kernel: messages + coord weights + scatter ------------
__global__ __launch_bounds__(NWAVES * 32) void egnn_edge_kernel(
    const _Float16* __restrict__ h16, const float* __restrict__ pos,
    const int* __restrict__ eidx,
    const _Float16* __restrict__ W1aT, const _Float16* __restrict__ W1bT,
    const float* __restrict__ w1d, const float* __restrict__ b1,
    const _Float16* __restrict__ W2T, const float* __restrict__ b2,
    const _Float16* __restrict__ cW1T, const float* __restrict__ cb1,
    const float* __restrict__ cW2,
    float* __restrict__ agg, float* __restrict__ coord) {
    __shared__ __align__(16) _Float16 sA[NWAVES][16][HDIM];   // h_row, later m
    __shared__ __align__(16) _Float16 sB[NWAVES][16][HDIM];   // h_col, later t
    __shared__ __align__(16) _Float16 sC[NWAVES][16][HDIM];   // m1
    __shared__ float sRel[NWAVES][16][4];                     // rel xyz + dsq
    __shared__ int   sRC[NWAVES][16][2];                      // row, col

    const int wave = threadIdx.x >> 5;
    const int lane = threadIdx.x & 31;
    const int e0   = (blockIdx.x * NWAVES + wave) << 4;
    const int c16  = lane & 15;
    const int rb   = (lane >> 4) << 3;

    if (lane < 16) {
        int e = e0 + lane;
        int r = eidx[e];
        int c = eidx[N_EDGES + e];
        sRC[wave][lane][0] = r;
        sRC[wave][lane][1] = c;
        float rx = pos[(size_t)r * 3 + 0] - pos[(size_t)c * 3 + 0];
        float ry = pos[(size_t)r * 3 + 1] - pos[(size_t)c * 3 + 1];
        float rz = pos[(size_t)r * 3 + 2] - pos[(size_t)c * 3 + 2];
        float d  = rx * rx + ry * ry + rz * rz;
        d = fminf(fmaxf(d, 1e-6f), 1e6f);
        sRel[wave][lane][0] = rx;
        sRel[wave][lane][1] = ry;
        sRel[wave][lane][2] = rz;
        sRel[wave][lane][3] = d;
    }
    __builtin_amdgcn_wave_barrier();

    // gather h[row] -> sA, h[col] -> sB (16B chunks)
    for (int idx = lane; idx < 256; idx += 32) {
        int e  = idx >> 4;
        int ch = (idx & 15) << 3;
        int r = sRC[wave][e][0];
        int c = sRC[wave][e][1];
        *(uint4*)&sA[wave][e][ch] = *(const uint4*)(h16 + (size_t)r * HDIM + ch);
        *(uint4*)&sB[wave][e][ch] = *(const uint4*)(h16 + (size_t)c * HDIM + ch);
    }
    __builtin_amdgcn_wave_barrier();

    // GEMM1: m1 = SiLU(h_r@W1a + h_c@W1b + dsq*w1d + b1) -> sC
    for (int nt = 0; nt < 8; nt += 2) {
        v8f acc0, acc1;
        float bv0 = b1[nt * 16 + c16];
        float bv1 = b1[nt * 16 + 16 + c16];
        #pragma unroll
        for (int i = 0; i < 8; ++i) { acc0[i] = bv0; acc1[i] = bv1; }
        #pragma unroll
        for (int kt = 0; kt < 4; ++kt) {
            v16h ar = load_A_lds(&sA[wave][0][0], lane, kt * 32, HDIM);
            v16h ac = load_A_lds(&sB[wave][0][0], lane, kt * 32, HDIM);
            v16h b0a = load_B_glb(W1aT, lane, nt * 16,      kt * 32, HDIM);
            v16h b1a = load_B_glb(W1aT, lane, nt * 16 + 16, kt * 32, HDIM);
            v16h b0b = load_B_glb(W1bT, lane, nt * 16,      kt * 32, HDIM);
            v16h b1b = load_B_glb(W1bT, lane, nt * 16 + 16, kt * 32, HDIM);
            acc0 = __builtin_amdgcn_wmma_f32_16x16x32_f16(false, ar, false, b0a, (short)0, acc0, false, false);
            acc1 = __builtin_amdgcn_wmma_f32_16x16x32_f16(false, ar, false, b1a, (short)0, acc1, false, false);
            acc0 = __builtin_amdgcn_wmma_f32_16x16x32_f16(false, ac, false, b0b, (short)0, acc0, false, false);
            acc1 = __builtin_amdgcn_wmma_f32_16x16x32_f16(false, ac, false, b1b, (short)0, acc1, false, false);
        }
        float wd0 = w1d[nt * 16 + c16];
        float wd1 = w1d[nt * 16 + 16 + c16];
        #pragma unroll
        for (int i = 0; i < 8; ++i) {
            float dsq = sRel[wave][rb + i][3];
            sC[wave][rb + i][nt * 16 + c16]      = (_Float16)silu_f(acc0[i] + dsq * wd0);
            sC[wave][rb + i][nt * 16 + 16 + c16] = (_Float16)silu_f(acc1[i] + dsq * wd1);
        }
    }
    __builtin_amdgcn_wave_barrier();

    // GEMM2: m = clip(SiLU(m1@W2 + b2), +-10) -> sA
    for (int nt = 0; nt < 8; nt += 2) {
        v8f acc0, acc1;
        float bv0 = b2[nt * 16 + c16];
        float bv1 = b2[nt * 16 + 16 + c16];
        #pragma unroll
        for (int i = 0; i < 8; ++i) { acc0[i] = bv0; acc1[i] = bv1; }
        #pragma unroll
        for (int kt = 0; kt < 4; ++kt) {
            v16h a  = load_A_lds(&sC[wave][0][0], lane, kt * 32, HDIM);
            v16h b0 = load_B_glb(W2T, lane, nt * 16,      kt * 32, HDIM);
            v16h b1v = load_B_glb(W2T, lane, nt * 16 + 16, kt * 32, HDIM);
            acc0 = __builtin_amdgcn_wmma_f32_16x16x32_f16(false, a, false, b0,  (short)0, acc0, false, false);
            acc1 = __builtin_amdgcn_wmma_f32_16x16x32_f16(false, a, false, b1v, (short)0, acc1, false, false);
        }
        #pragma unroll
        for (int i = 0; i < 8; ++i) {
            float x0 = fminf(fmaxf(silu_f(acc0[i]), -10.0f), 10.0f);
            float x1 = fminf(fmaxf(silu_f(acc1[i]), -10.0f), 10.0f);
            sA[wave][rb + i][nt * 16 + c16]      = (_Float16)x0;
            sA[wave][rb + i][nt * 16 + 16 + c16] = (_Float16)x1;
        }
    }
    __builtin_amdgcn_wave_barrier();

    // GEMM3: t = SiLU(m@cW1 + cb1) -> sB
    for (int nt = 0; nt < 8; nt += 2) {
        v8f acc0, acc1;
        float bv0 = cb1[nt * 16 + c16];
        float bv1 = cb1[nt * 16 + 16 + c16];
        #pragma unroll
        for (int i = 0; i < 8; ++i) { acc0[i] = bv0; acc1[i] = bv1; }
        #pragma unroll
        for (int kt = 0; kt < 4; ++kt) {
            v16h a  = load_A_lds(&sA[wave][0][0], lane, kt * 32, HDIM);
            v16h b0 = load_B_glb(cW1T, lane, nt * 16,      kt * 32, HDIM);
            v16h b1v = load_B_glb(cW1T, lane, nt * 16 + 16, kt * 32, HDIM);
            acc0 = __builtin_amdgcn_wmma_f32_16x16x32_f16(false, a, false, b0,  (short)0, acc0, false, false);
            acc1 = __builtin_amdgcn_wmma_f32_16x16x32_f16(false, a, false, b1v, (short)0, acc1, false, false);
        }
        #pragma unroll
        for (int i = 0; i < 8; ++i) {
            sB[wave][rb + i][nt * 16 + c16]      = (_Float16)silu_f(acc0[i]);
            sB[wave][rb + i][nt * 16 + 16 + c16] = (_Float16)silu_f(acc1[i]);
        }
    }
    __builtin_amdgcn_wave_barrier();

    // cw = t . cW2 ; scatter coord update + neighbor count
    if (lane < 16) {
        float s = 0.f;
        for (int j = 0; j < HDIM; ++j)
            s += (float)sB[wave][lane][j] * cW2[j];
        int r = sRC[wave][lane][0];
        atomicAdd(&coord[(size_t)r * 4 + 0], s * sRel[wave][lane][0]);
        atomicAdd(&coord[(size_t)r * 4 + 1], s * sRel[wave][lane][1]);
        atomicAdd(&coord[(size_t)r * 4 + 2], s * sRel[wave][lane][2]);
        atomicAdd(&coord[(size_t)r * 4 + 3], 1.0f);
    }
    // scatter messages into agg
    for (int e = 0; e < 16; ++e) {
        int r = sRC[wave][e][0];
        #pragma unroll
        for (int jj = 0; jj < 4; ++jj) {
            int j = lane + jj * 32;
            atomicAdd(&agg[(size_t)r * HDIM + j], (float)sA[wave][e][j]);
        }
    }
}

// ---------------- node kernel: pos update + node MLP + residual LN -----------
__global__ __launch_bounds__(NWAVES * 32) void egnn_node_kernel(
    float* __restrict__ h32, _Float16* __restrict__ h16,
    const float* __restrict__ agg,
    const float* __restrict__ posin, const float* __restrict__ coord,
    float* __restrict__ posout,
    const _Float16* __restrict__ nW1aT, const _Float16* __restrict__ nW1bT,
    const float* __restrict__ nb1,
    const _Float16* __restrict__ nW2T, const float* __restrict__ nb2,
    const float* __restrict__ lng, const float* __restrict__ lnb) {
    __shared__ __align__(16) _Float16 sA[NWAVES][16][HDIM];
    __shared__ __align__(16) _Float16 sB[NWAVES][16][HDIM];
    __shared__ __align__(16) _Float16 sC[NWAVES][16][HDIM];

    const int wave = threadIdx.x >> 5;
    const int lane = threadIdx.x & 31;
    const int n0   = (blockIdx.x * NWAVES + wave) << 4;
    const int c16  = lane & 15;
    const int rb   = (lane >> 4) << 3;

    // pos_new = pos + coord_update / (num_neighbors + 1e-6)
    if (lane < 16) {
        int n = n0 + lane;
        float inv = 1.0f / (coord[(size_t)n * 4 + 3] + 1e-6f);
        posout[(size_t)n * 3 + 0] = posin[(size_t)n * 3 + 0] + coord[(size_t)n * 4 + 0] * inv;
        posout[(size_t)n * 3 + 1] = posin[(size_t)n * 3 + 1] + coord[(size_t)n * 4 + 1] * inv;
        posout[(size_t)n * 3 + 2] = posin[(size_t)n * 3 + 2] + coord[(size_t)n * 4 + 2] * inv;
    }

    // stage h (f16) and agg (f32 -> f16)
    for (int idx = lane; idx < 256; idx += 32) {
        int r  = idx >> 4;
        int ch = (idx & 15) << 3;
        int n = n0 + r;
        *(uint4*)&sA[wave][r][ch] = *(const uint4*)(h16 + (size_t)n * HDIM + ch);
        #pragma unroll
        for (int j = 0; j < 8; ++j)
            sB[wave][r][ch + j] = (_Float16)agg[(size_t)n * HDIM + ch + j];
    }
    __builtin_amdgcn_wave_barrier();

    // mid = SiLU(h@nW1a + agg@nW1b + nb1) -> sC
    for (int nt = 0; nt < 8; nt += 2) {
        v8f acc0, acc1;
        float bv0 = nb1[nt * 16 + c16];
        float bv1 = nb1[nt * 16 + 16 + c16];
        #pragma unroll
        for (int i = 0; i < 8; ++i) { acc0[i] = bv0; acc1[i] = bv1; }
        #pragma unroll
        for (int kt = 0; kt < 4; ++kt) {
            v16h ah = load_A_lds(&sA[wave][0][0], lane, kt * 32, HDIM);
            v16h ag = load_A_lds(&sB[wave][0][0], lane, kt * 32, HDIM);
            v16h b0a = load_B_glb(nW1aT, lane, nt * 16,      kt * 32, HDIM);
            v16h b1a = load_B_glb(nW1aT, lane, nt * 16 + 16, kt * 32, HDIM);
            v16h b0b = load_B_glb(nW1bT, lane, nt * 16,      kt * 32, HDIM);
            v16h b1b = load_B_glb(nW1bT, lane, nt * 16 + 16, kt * 32, HDIM);
            acc0 = __builtin_amdgcn_wmma_f32_16x16x32_f16(false, ah, false, b0a, (short)0, acc0, false, false);
            acc1 = __builtin_amdgcn_wmma_f32_16x16x32_f16(false, ah, false, b1a, (short)0, acc1, false, false);
            acc0 = __builtin_amdgcn_wmma_f32_16x16x32_f16(false, ag, false, b0b, (short)0, acc0, false, false);
            acc1 = __builtin_amdgcn_wmma_f32_16x16x32_f16(false, ag, false, b1b, (short)0, acc1, false, false);
        }
        #pragma unroll
        for (int i = 0; i < 8; ++i) {
            sC[wave][rb + i][nt * 16 + c16]      = (_Float16)silu_f(acc0[i]);
            sC[wave][rb + i][nt * 16 + 16 + c16] = (_Float16)silu_f(acc1[i]);
        }
    }
    __builtin_amdgcn_wave_barrier();

    // u = mid@nW2 + nb2, stored f32 into the (dead) sA/sB area
    for (int nt = 0; nt < 8; nt += 2) {
        v8f acc0, acc1;
        float bv0 = nb2[nt * 16 + c16];
        float bv1 = nb2[nt * 16 + 16 + c16];
        #pragma unroll
        for (int i = 0; i < 8; ++i) { acc0[i] = bv0; acc1[i] = bv1; }
        #pragma unroll
        for (int kt = 0; kt < 4; ++kt) {
            v16h a  = load_A_lds(&sC[wave][0][0], lane, kt * 32, HDIM);
            v16h b0 = load_B_glb(nW2T, lane, nt * 16,      kt * 32, HDIM);
            v16h b1v = load_B_glb(nW2T, lane, nt * 16 + 16, kt * 32, HDIM);
            acc0 = __builtin_amdgcn_wmma_f32_16x16x32_f16(false, a, false, b0,  (short)0, acc0, false, false);
            acc1 = __builtin_amdgcn_wmma_f32_16x16x32_f16(false, a, false, b1v, (short)0, acc1, false, false);
        }
        #pragma unroll
        for (int i = 0; i < 8; ++i) {
            int row = rb + i;
            int col0 = nt * 16 + c16, col1 = col0 + 16;
            float* d0 = (col0 < 64) ? ((float*)&sA[wave][row][0]) + col0
                                    : ((float*)&sB[wave][row][0]) + (col0 - 64);
            float* d1 = (col1 < 64) ? ((float*)&sA[wave][row][0]) + col1
                                    : ((float*)&sB[wave][row][0]) + (col1 - 64);
            *d0 = acc0[i];
            *d1 = acc1[i];
        }
    }
    __builtin_amdgcn_wave_barrier();

    // residual + LayerNorm (one node per lane)
    if (lane < 16) {
        int n = n0 + lane;
        float* u0 = (float*)&sA[wave][lane][0];
        float* u1 = (float*)&sB[wave][lane][0];
        float mu = 0.f;
        for (int j = 0; j < 64; ++j) { float v = h32[(size_t)n * HDIM + j]      + u0[j]; u0[j] = v; mu += v; }
        for (int j = 0; j < 64; ++j) { float v = h32[(size_t)n * HDIM + 64 + j] + u1[j]; u1[j] = v; mu += v; }
        mu *= (1.0f / 128.0f);
        float var = 0.f;
        for (int j = 0; j < 64; ++j) { float d = u0[j] - mu; var += d * d; }
        for (int j = 0; j < 64; ++j) { float d = u1[j] - mu; var += d * d; }
        var *= (1.0f / 128.0f);
        float rs = rsqrtf(var + 1e-5f);
        for (int j = 0; j < HDIM; ++j) {
            float v = (j < 64) ? u0[j] : u1[j - 64];
            float y = (v - mu) * rs * lng[j] + lnb[j];
            h32[(size_t)n * HDIM + j] = y;
            h16[(size_t)n * HDIM + j] = (_Float16)y;
        }
    }
}

// =============================================================================
extern "C" void kernel_launch(void* const* d_in, const int* in_sizes, int n_in,
                              void* d_out, int out_size, void* d_ws, size_t ws_size,
                              hipStream_t stream) {
    const float* z     = (const float*)d_in[0];
    const float* atom  = (const float*)d_in[1];
    const float* pos0  = (const float*)d_in[2];
    const int*   eidx  = (const int*)d_in[3];
    const float* injW1 = (const float*)d_in[4];
    const float* injb1 = (const float*)d_in[5];
    const float* injW2 = (const float*)d_in[6];
    const float* injb2 = (const float*)d_in[7];
    const float* injW3 = (const float*)d_in[8];
    const float* injb3 = (const float*)d_in[9];
    const float* eW1   = (const float*)d_in[10];
    const float* eb1   = (const float*)d_in[11];
    const float* eW2   = (const float*)d_in[12];
    const float* eb2   = (const float*)d_in[13];
    const float* cW1   = (const float*)d_in[14];
    const float* cb1   = (const float*)d_in[15];
    const float* cW2   = (const float*)d_in[16];
    const float* nW1   = (const float*)d_in[17];
    const float* nb1   = (const float*)d_in[18];
    const float* nW2   = (const float*)d_in[19];
    const float* nb2   = (const float*)d_in[20];
    const float* lng   = (const float*)d_in[21];
    const float* lnb   = (const float*)d_in[22];

    char* ws = (char*)d_ws;
    size_t off = 0;
    auto carve = [&](size_t bytes) -> char* {
        char* p = ws + off;
        off = (off + bytes + 255) & ~(size_t)255;
        return p;
    };
    const size_t WSZ = (size_t)HDIM * HDIM;             // 16384 elements
    _Float16* WtI1  = (_Float16*)carve(256 * 96 * 2);
    _Float16* WtI2  = (_Float16*)carve(128 * 256 * 2);
    _Float16* WtI3  = (_Float16*)carve(WSZ * 2);
    _Float16* W1aT  = (_Float16*)carve(4 * WSZ * 2);
    _Float16* W1bT  = (_Float16*)carve(4 * WSZ * 2);
    _Float16* W2T   = (_Float16*)carve(4 * WSZ * 2);
    _Float16* cW1T  = (_Float16*)carve(4 * WSZ * 2);
    _Float16* nW1aT = (_Float16*)carve(4 * WSZ * 2);
    _Float16* nW1bT = (_Float16*)carve(4 * WSZ * 2);
    _Float16* nW2T  = (_Float16*)carve(4 * WSZ * 2);
    float*    h32   = (float*)   carve((size_t)N_NODES * HDIM * 4);
    _Float16* h16   = (_Float16*)carve((size_t)N_NODES * HDIM * 2);
    float*    agg   = (float*)   carve((size_t)N_NODES * HDIM * 4);
    float*    coord = (float*)   carve((size_t)N_NODES * 4 * 4);
    float*    posA  = (float*)   carve((size_t)N_NODES * 3 * 4);
    float*    posB  = (float*)   carve((size_t)N_NODES * 3 * 4);
    (void)in_sizes; (void)n_in; (void)out_size; (void)ws_size;

    // ---- weight prep (all tiny; L2-resident thereafter) ----
    const int TB = 256;
    transpose_pad_f16_kernel<<<(256 * 96 + TB - 1) / TB, TB, 0, stream>>>(WtI1, injW1, 256, 96, 80, 256);
    transpose_f16_kernel<<<(128 * 256 + TB - 1) / TB, TB, 0, stream>>>(WtI2, injW2, 128, 256, 128);
    transpose_f16_kernel<<<((int)WSZ + TB - 1) / TB, TB, 0, stream>>>(WtI3, injW3, 128, 128, 128);
    for (int l = 0; l < 4; ++l) {
        transpose_f16_kernel<<<((int)WSZ + TB - 1) / TB, TB, 0, stream>>>(
            W1aT + l * WSZ, eW1 + (size_t)l * 257 * 128, 128, 128, 128);
        transpose_f16_kernel<<<((int)WSZ + TB - 1) / TB, TB, 0, stream>>>(
            W1bT + l * WSZ, eW1 + (size_t)l * 257 * 128 + 128 * 128, 128, 128, 128);
        transpose_f16_kernel<<<((int)WSZ + TB - 1) / TB, TB, 0, stream>>>(
            W2T + l * WSZ, eW2 + l * WSZ, 128, 128, 128);
        transpose_f16_kernel<<<((int)WSZ + TB - 1) / TB, TB, 0, stream>>>(
            cW1T + l * WSZ, cW1 + l * WSZ, 128, 128, 128);
        transpose_f16_kernel<<<((int)WSZ + TB - 1) / TB, TB, 0, stream>>>(
            nW1aT + l * WSZ, nW1 + (size_t)l * 256 * 128, 128, 128, 128);
        transpose_f16_kernel<<<((int)WSZ + TB - 1) / TB, TB, 0, stream>>>(
            nW1bT + l * WSZ, nW1 + (size_t)l * 256 * 128 + 128 * 128, 128, 128, 128);
        transpose_f16_kernel<<<((int)WSZ + TB - 1) / TB, TB, 0, stream>>>(
            nW2T + l * WSZ, nW2 + l * WSZ, 128, 128, 128);
    }

    // ---- latent injection ----
    egnn_inject_kernel<<<N_NODES / (NWAVES * 16), NWAVES * 32, 0, stream>>>(
        z, atom, WtI1, injb1, WtI2, injb2, WtI3, injb3, h32, h16);

    // ---- 4 EGNN layers ----
    const float* pin = pos0;
    float* pbuf[2] = {posA, posB};
    for (int l = 0; l < 4; ++l) {
        hipMemsetAsync(agg, 0, (size_t)N_NODES * HDIM * 4, stream);
        hipMemsetAsync(coord, 0, (size_t)N_NODES * 4 * 4, stream);
        egnn_edge_kernel<<<N_EDGES / (NWAVES * 16), NWAVES * 32, 0, stream>>>(
            h16, pin, eidx,
            W1aT + l * WSZ, W1bT + l * WSZ,
            eW1 + (size_t)l * 257 * 128 + 256 * 128, eb1 + (size_t)l * 128,
            W2T + l * WSZ, eb2 + (size_t)l * 128,
            cW1T + l * WSZ, cb1 + (size_t)l * 128, cW2 + (size_t)l * 128,
            agg, coord);
        float* pout = (l == 3) ? (float*)d_out : pbuf[l & 1];
        egnn_node_kernel<<<N_NODES / (NWAVES * 16), NWAVES * 32, 0, stream>>>(
            h32, h16, agg, pin, coord, pout,
            nW1aT + l * WSZ, nW1bT + l * WSZ, nb1 + (size_t)l * 128,
            nW2T + l * WSZ, nb2 + (size_t)l * 128,
            lng + (size_t)l * 128, lnb + (size_t)l * 128);
        pin = pout;
    }
}
